// BasicGraphConvNet_4741643895543
// MI455X (gfx1250) — compile-verified, matrix operand order checked by reading, not run on hardware
//
#include <hip/hip_runtime.h>

#define N_NODES 50000
#define N_EDGES 800000
#define NUM_GRAPHS 4
#define IN_DIM 1024
#define HID 128
#define MAX_RISK 5.0f

typedef __attribute__((ext_vector_type(16))) __bf16 v16bf;
typedef __attribute__((ext_vector_type(8)))  float  v8f;
typedef __attribute__((ext_vector_type(4)))  float  v4f;
typedef __attribute__((ext_vector_type(4)))  int    v4i;

#if defined(__has_builtin)
#if __has_builtin(__builtin_amdgcn_global_load_async_to_lds_b128)
#define USE_ASYNC_LDS 1
#endif
#endif

union BF16Frag {
    v16bf v;
    unsigned u[8];
    uint4 q[2];
};

// pack two f32 -> one dword of two bf16 (lo, hi), round-to-nearest-even
__device__ __forceinline__ unsigned pack_bf16x2(float lo, float hi) {
#if defined(__has_builtin) && __has_builtin(__builtin_amdgcn_cvt_pk_bf16_f32)
    typedef __attribute__((ext_vector_type(2))) __bf16 v2bf;
    v2bf p = __builtin_amdgcn_cvt_pk_bf16_f32(lo, hi);
    return __builtin_bit_cast(unsigned, p);
#else
    unsigned a = __float_as_uint(lo), b = __float_as_uint(hi);
    a += 0x7fffu + ((a >> 16) & 1u);
    b += 0x7fffu + ((b >> 16) & 1u);
    return (a >> 16) | (b & 0xffff0000u);
#endif
}

template <int N>
__device__ __forceinline__ void wait_asynccnt() {
#if defined(__has_builtin) && __has_builtin(__builtin_amdgcn_s_wait_asynccnt)
    __builtin_amdgcn_s_wait_asynccnt(N);
#else
    asm volatile("s_wait_asynccnt %0" ::"i"(N) : "memory");
#endif
}

// ---------------------------------------------------------------------------
// Pre-convert W[K x 128] f32 -> packed bf16 panels. Panel for K-step s is an
// 8KB block whose byte image equals the LDS B layout: uint at [n*16 + k/2],
// dword = (bf16(W[k+1][n]) << 16) | bf16(W[k][n]).
// ---------------------------------------------------------------------------
__global__ void convert_w(const float* __restrict__ W, unsigned* __restrict__ Wbf,
                          int K) {
    int idx = blockIdx.x * blockDim.x + threadIdx.x;
    if (idx >= K * 64) return;
    int blk = idx >> 11;        // K-step
    int r   = idx & 2047;
    int nn  = r >> 4;           // column 0..127
    int kp  = r & 15;           // k-pair 0..15
    int k0  = blk * 32 + kp * 2;
    Wbf[idx] = pack_bf16x2(W[(long)k0 * HID + nn], W[(long)(k0 + 1) * HID + nn]);
}

// ---------------------------------------------------------------------------
// Y[n x 128] = X[n x K] @ W[K x 128], bf16 x bf16 -> f32 WMMA.
// Block = 256 threads = 8 waves computing 128 rows x 128 cols.
// B panels stream global->LDS (async, double-buffered when available).
// OOB rows clamp to row 0: D row m depends only on A row m, and OOB rows are
// masked at the store, so no per-iteration bounds branch is needed.
// ---------------------------------------------------------------------------
__global__ __launch_bounds__(256)
void gemm_xw_bf16(const float* __restrict__ X, const unsigned* __restrict__ Wbf,
                  float* __restrict__ Y, int n, int K) {
#ifdef USE_ASYNC_LDS
    __shared__ unsigned ldsB[2][2048];      // 2 x 8KB panels
#else
    __shared__ unsigned ldsB[1][2048];
#endif
    const int tid  = threadIdx.x;
    const int wave = tid >> 5;
    const int lane = tid & 31;
    const int half = lane >> 4;             // 0/1
    const int mrow = lane & 15;             // A row / B-C col within tile
    const int rowBase = blockIdx.x * 128 + wave * 16;
    const int arow = rowBase + mrow;
    const float* xrow = X + (long)(arow < n ? arow : 0) * K;   // clamp, no branch
    const int steps = K >> 5;

    v8f acc[8];
    #pragma unroll
    for (int t = 0; t < 8; ++t) acc[t] = (v8f){};

#ifdef USE_ASYNC_LDS
    auto issue_panel = [&](int s, int buf) {
        v4i* g = (v4i*)(Wbf + (long)s * 2048) + tid;   // global source
        v4i* l = (v4i*)&ldsB[buf][0] + tid;            // LDS destination
        __builtin_amdgcn_global_load_async_to_lds_b128(g, l, 0, 0);
        __builtin_amdgcn_global_load_async_to_lds_b128(g + 256, l + 256, 0, 0);
    };
    issue_panel(0, 0);      // prologue: panel for step 0
#endif

    for (int s = 0; s < steps; ++s) {
        const int kb = s << 5;

        // ---- A fragment: lane half h owns K = {8h..8h+7} U {16+8h..16+8h+7}
        // X is streamed exactly once -> non-temporal loads keep L2 for the
        // gather/scatter working set.
        BF16Frag a;
        {
            const v4f* pa = (const v4f*)(xrow + kb + half * 8);
            v4f f0 = __builtin_nontemporal_load(pa);
            v4f f1 = __builtin_nontemporal_load(pa + 1);
            v4f f2 = __builtin_nontemporal_load(pa + 4);
            v4f f3 = __builtin_nontemporal_load(pa + 5);
            a.u[0] = pack_bf16x2(f0[0], f0[1]); a.u[1] = pack_bf16x2(f0[2], f0[3]);
            a.u[2] = pack_bf16x2(f1[0], f1[1]); a.u[3] = pack_bf16x2(f1[2], f1[3]);
            a.u[4] = pack_bf16x2(f2[0], f2[1]); a.u[5] = pack_bf16x2(f2[2], f2[3]);
            a.u[6] = pack_bf16x2(f3[0], f3[1]); a.u[7] = pack_bf16x2(f3[2], f3[3]);
        }

#ifdef USE_ASYNC_LDS
        const int buf = s & 1;
        __syncthreads();                        // all waves done reading buf^1
        if (s + 1 < steps) {
            issue_panel(s + 1, buf ^ 1);        // overlap next panel with compute
            wait_asynccnt<2>();                 // step-s panel (oldest 2) arrived
        } else {
            wait_asynccnt<0>();
        }
        __syncthreads();                        // every wave's panel loads visible
#else
        const int buf = 0;
        __syncthreads();
        {   // synchronous 8KB panel copy, conversion-free
            const uint4* g = (const uint4*)(Wbf + (long)s * 2048);
            uint4* l = (uint4*)&ldsB[0][0];
            l[tid]       = g[tid];
            l[tid + 256] = g[tid + 256];
        }
        __syncthreads();
#endif

        // ---- load all 8 B fragments first so DS loads pipeline, then 8
        // back-to-back WMMAs (no D->A/B hazard: distinct accumulators).
        BF16Frag bfr[8];
        #pragma unroll
        for (int t = 0; t < 8; ++t) {
            const uint4* pb =
                (const uint4*)&ldsB[buf][(t * 16 + mrow) * 16 + 4 * half];
            bfr[t].q[0] = pb[0];    // K = 8h .. 8h+7
            bfr[t].q[1] = pb[2];    // K = 16+8h .. 16+8h+7
        }
        #pragma unroll
        for (int t = 0; t < 8; ++t) {
            acc[t] = __builtin_amdgcn_wmma_f32_16x16x32_bf16(
                false, a.v, false, bfr[t].v, (short)0, acc[t], false, false);
        }
    }

    // ---- store: C/D layout -> VGPR r: lanes0-15 M=r, lanes16-31 M=8+r ----
    #pragma unroll
    for (int t = 0; t < 8; ++t) {
        const int col = t * 16 + mrow;
        #pragma unroll
        for (int r = 0; r < 8; ++r) {
            const int orow = rowBase + r + 8 * half;
            if (orow < n) Y[(long)orow * HID + col] = acc[t][r];
        }
    }
}

// ---------------------------------------------------------------------------
__global__ void fill_f32(float* __restrict__ p, float v, long n) {
    long i = (long)blockIdx.x * blockDim.x + threadIdx.x;
    if (i < n) p[i] = v;
}

__global__ void fill4_f32(float4* __restrict__ p, float v, long n4) {
    long i = (long)blockIdx.x * blockDim.x + threadIdx.x;
    if (i < n4) p[i] = make_float4(v, v, v, v);
}

__global__ void deg_accum(const int* __restrict__ dst, float* __restrict__ deg, int E) {
    int e = blockIdx.x * blockDim.x + threadIdx.x;
    if (e < E) atomicAdd(&deg[dst[e]], 1.0f);
}

__global__ void rsqrt_k(const float* __restrict__ deg, float* __restrict__ dinv, int n) {
    int i = blockIdx.x * blockDim.x + threadIdx.x;
    if (i < n) {
        float d = deg[i];
        dinv[i] = d > 0.0f ? rsqrtf(d) : 0.0f;
    }
}

// One wave per edge; each lane handles 4 of 128 columns.
__global__ __launch_bounds__(256)
void scatter_edges(const float* __restrict__ xw, const int* __restrict__ src,
                   const int* __restrict__ dst, const float* __restrict__ dinv,
                   float* __restrict__ agg, int E) {
    long idx = (long)blockIdx.x * blockDim.x + threadIdx.x;
    int e    = (int)(idx >> 5);
    int lane = (int)(idx & 31);
    if (e >= E) return;
    int s = src[e], d = dst[e];
    float nrm = dinv[s] * dinv[d];
    float4 v = *(const float4*)(xw + ((long)s << 7) + lane * 4);
    float* po = agg + ((long)d << 7) + lane * 4;
    atomicAdd(po + 0, v.x * nrm);
    atomicAdd(po + 1, v.y * nrm);
    atomicAdd(po + 2, v.z * nrm);
    atomicAdd(po + 3, v.w * nrm);
}

// h = relu(agg + xw * dinv^2 (self loop) + bias), in place over agg. float4 lanes.
__global__ void combine_relu(float* __restrict__ hb, const float* __restrict__ xw,
                             const float* __restrict__ dinv,
                             const float* __restrict__ bias, int n) {
    long idx = (long)blockIdx.x * blockDim.x + threadIdx.x;
    if (idx >= (long)n * 32) return;
    int i  = (int)(idx >> 5);
    int c4 = (int)(idx & 31) << 2;
    float di = dinv[i];
    float sl = di * di;
    long base = ((long)i << 7) + c4;
    float4 xv = *(const float4*)(xw + base);
    float4 hv = *(const float4*)(hb + base);
    float4 bv = *(const float4*)(bias + c4);
    float4 r;
    r.x = hv.x + xv.x * sl + bv.x; r.x = r.x > 0.0f ? r.x : 0.0f;
    r.y = hv.y + xv.y * sl + bv.y; r.y = r.y > 0.0f ? r.y : 0.0f;
    r.z = hv.z + xv.z * sl + bv.z; r.z = r.z > 0.0f ? r.z : 0.0f;
    r.w = hv.w + xv.w * sl + bv.w; r.w = r.w > 0.0f ? r.w : 0.0f;
    *(float4*)(hb + base) = r;
}

// Global max (uint bit-trick: h >= 0 post-ReLU) + sum + count per graph.
__global__ void pool_kernel(const float* __restrict__ h, const int* __restrict__ batch,
                            unsigned* __restrict__ gmaxU, float* __restrict__ gsum,
                            float* __restrict__ cnt, int n) {
    long idx = (long)blockIdx.x * blockDim.x + threadIdx.x;
    if (idx >= (long)n * 32) return;
    int i  = (int)(idx >> 5);
    int c4 = (int)(idx & 31) << 2;
    int g = batch[i];
    float4 v = *(const float4*)(h + ((long)i << 7) + c4);
    unsigned* pm = gmaxU + g * HID + c4;
    float*    ps = gsum  + g * HID + c4;
    atomicMax(pm + 0, __float_as_uint(v.x)); atomicAdd(ps + 0, v.x);
    atomicMax(pm + 1, __float_as_uint(v.y)); atomicAdd(ps + 1, v.y);
    atomicMax(pm + 2, __float_as_uint(v.z)); atomicAdd(ps + 2, v.z);
    atomicMax(pm + 3, __float_as_uint(v.w)); atomicAdd(ps + 3, v.w);
    if (c4 == 0) atomicAdd(&cnt[g], 1.0f);
}

// Tiny MLP head: [4,256] -> relu 128 -> relu 64 -> 1, clamp at MAX_RISK.
__global__ __launch_bounds__(256)
void mlp_kernel(const unsigned* __restrict__ gmaxU, const float* __restrict__ gsum,
                const float* __restrict__ cnt,
                const float* __restrict__ Wl1, const float* __restrict__ bl1,
                const float* __restrict__ Wl2, const float* __restrict__ bl2,
                const float* __restrict__ Wl3, const float* __restrict__ bl3,
                float* __restrict__ out) {
    __shared__ float g [NUM_GRAPHS][2 * HID];
    __shared__ float g1[NUM_GRAPHS][HID];
    __shared__ float g2[NUM_GRAPHS][HID / 2];
    const int t = threadIdx.x;

    for (int k = t; k < NUM_GRAPHS * 2 * HID; k += 256) {
        int gi = k / (2 * HID), c = k % (2 * HID);
        float v;
        if (c < HID) {
            v = __uint_as_float(gmaxU[gi * HID + c]);
        } else {
            float cc = cnt[gi];
            cc = cc > 1.0f ? cc : 1.0f;
            v = gsum[gi * HID + (c - HID)] / cc;
        }
        g[gi][c] = v;
    }
    __syncthreads();

    for (int k = t; k < NUM_GRAPHS * HID; k += 256) {
        int gi = k / HID, o = k % HID;
        float s = bl1[o];
        for (int kk = 0; kk < 2 * HID; ++kk) s += g[gi][kk] * Wl1[kk * HID + o];
        g1[gi][o] = s > 0.0f ? s : 0.0f;
    }
    __syncthreads();

    if (t < NUM_GRAPHS * (HID / 2)) {
        int gi = t / (HID / 2), o = t % (HID / 2);
        float s = bl2[o];
        for (int kk = 0; kk < HID; ++kk) s += g1[gi][kk] * Wl2[kk * (HID / 2) + o];
        g2[gi][o] = s > 0.0f ? s : 0.0f;
    }
    __syncthreads();

    if (t < NUM_GRAPHS) {
        float s = bl3[0];
        for (int kk = 0; kk < HID / 2; ++kk) s += g2[t][kk] * Wl3[kk];
        out[t] = s < MAX_RISK ? s : MAX_RISK;
    }
}

// ---------------------------------------------------------------------------
extern "C" void kernel_launch(void* const* d_in, const int* in_sizes, int n_in,
                              void* d_out, int out_size, void* d_ws, size_t ws_size,
                              hipStream_t stream) {
    (void)in_sizes; (void)n_in; (void)out_size; (void)ws_size;

    const float* x     = (const float*)d_in[0];
    const int*   ei    = (const int*)d_in[1];
    const int*   batch = (const int*)d_in[2];
    const float* W0  = (const float*)d_in[3];  const float* b0  = (const float*)d_in[4];
    const float* W1  = (const float*)d_in[5];  const float* b1  = (const float*)d_in[6];
    const float* W2  = (const float*)d_in[7];  const float* b2  = (const float*)d_in[8];
    const float* Wl1 = (const float*)d_in[9];  const float* bl1 = (const float*)d_in[10];
    const float* Wl2 = (const float*)d_in[11]; const float* bl2 = (const float*)d_in[12];
    const float* Wl3 = (const float*)d_in[13]; const float* bl3 = (const float*)d_in[14];
    const int* src = ei;
    const int* dst = ei + N_EDGES;

    // workspace layout (floats; offsets keep 16B alignment)
    float* xw   = (float*)d_ws;                       // N*128
    float* hb   = xw + (long)N_NODES * HID;           // N*128 (agg, then h in place)
    float* deg  = hb + (long)N_NODES * HID;           // N
    float* dinv = deg + N_NODES;                      // N
    float* gmax = dinv + N_NODES;                     // 4*128 (uint-punned)
    float* gsum = gmax + NUM_GRAPHS * HID;            // 4*128
    float* cnt  = gsum + NUM_GRAPHS * HID;            // 4
    unsigned* wbf = (unsigned*)(cnt + NUM_GRAPHS + 12); // K*64 dwords (<=256KB), 16B aligned

    const long nfeat = (long)N_NODES * HID;

    // degrees (self-loop contributes 1 -> init deg = 1)
    fill_f32<<<(N_NODES + 255) / 256, 256, 0, stream>>>(deg, 1.0f, N_NODES);
    deg_accum<<<(N_EDGES + 255) / 256, 256, 0, stream>>>(dst, deg, N_EDGES);
    rsqrt_k<<<(N_NODES + 255) / 256, 256, 0, stream>>>(deg, dinv, N_NODES);

    const int gemmGrid = (N_NODES + 127) / 128;
    const float* Ws[3] = {W0, W1, W2};
    const float* bs[3] = {b0, b1, b2};
    const float* Xin = x;
    int K = IN_DIM;
    for (int l = 0; l < 3; ++l) {
        convert_w<<<(K * 64 + 255) / 256, 256, 0, stream>>>(Ws[l], wbf, K);
        gemm_xw_bf16<<<gemmGrid, 256, 0, stream>>>(Xin, wbf, xw, N_NODES, K);
        fill4_f32<<<(int)((nfeat / 4 + 255) / 256), 256, 0, stream>>>(
            (float4*)hb, 0.0f, nfeat / 4);
        scatter_edges<<<(int)(((long)N_EDGES * 32 + 255) / 256), 256, 0, stream>>>(
            xw, src, dst, dinv, hb, N_EDGES);
        combine_relu<<<(int)(((long)N_NODES * 32 + 255) / 256), 256, 0, stream>>>(
            hb, xw, dinv, bs[l], N_NODES);
        Xin = hb;
        K = HID;
    }

    // pooling
    fill_f32<<<(2 * NUM_GRAPHS * HID + NUM_GRAPHS + 255) / 256, 256, 0, stream>>>(
        gmax, 0.0f, 2 * NUM_GRAPHS * HID + NUM_GRAPHS);
    pool_kernel<<<(int)(((long)N_NODES * 32 + 255) / 256), 256, 0, stream>>>(
        hb, batch, (unsigned*)gmax, gsum, cnt, N_NODES);

    mlp_kernel<<<1, 256, 0, stream>>>((const unsigned*)gmax, gsum, cnt,
                                      Wl1, bl1, Wl2, bl2, Wl3, bl3, (float*)d_out);
}
